// Sequence_Modeling_9637906612878
// MI455X (gfx1250) — matrix operand order repaired and checked by reading, not compiled, not run on hardware
//
#include <hip/hip_runtime.h>

// ---------------------------------------------------------------------------
// Seq2seq RNN + attention for MI455X (gfx1250, wave32, WMMA).
//   V=32000 E=512 H=1024 O=32000  B=16 SE=128 SD=64
// Strategy:
//   * cast all weights + gathered embeddings to f16 (one pass)
//   * encoder: persistent 1-block kernel (512 thr = 16 waves), h in LDS,
//     WMMA f32<-f16 16x16x32, 4 N-tiles per wave with A-fragment reuse
//   * decoder recurrence: persistent 1-block kernel (q/scores/softmax/ctx/cell)
//   * output projection batched into ONE (1024 x 32000 x 1024) WMMA GEMM
//     (out_W f16 = 65MB fits the 192MB L2 -> compute bound, not HBM bound)
// Register budget: 512-thread blocks = 4 waves/SIMD -> 128 VGPR/lane; the
// per-wave live set (acc[4]=32 + A 8 + B 8..32 + addr) fits with no spill.
// K-loops kept rolled (#pragma unroll 1) so the compiler cannot preload
// whole K panels and spill to scratch. Epilogues use a branchless tanh
// (no EXEC-mask churn from libm tanhf).
// ---------------------------------------------------------------------------

typedef __attribute__((ext_vector_type(16))) _Float16 v16h;
typedef __attribute__((ext_vector_type(8)))  _Float16 v8h;
typedef __attribute__((ext_vector_type(8)))  float    v8f;

#define HDIM 1024
#define EDIM 512
#define ODIM 32000
#define BDIM 16
#define SEQE 128
#define SEQD 64
#define IDIM 1536   // E + H
#define RNN_THREADS 512   // 16 wave32 -> 4 waves/SIMD -> 128 VGPRs/lane

__device__ __forceinline__ v8f wmma16(v16h a, v16h b, v8f c) {
  // D = A(16x32 f16) * B(32x16 f16) + C(16x16 f32)
  return __builtin_amdgcn_wmma_f32_16x16x32_f16(
      /*neg_a=*/false, a, /*neg_b=*/false, b,
      /*c_mod=*/(short)0, c, /*reuse_a=*/false, /*reuse_b=*/false);
}

// Branchless tanh: 1 - 2/(e^{2x}+1).  e->inf => 1, e->0 => -1; no NaN, no
// divergent EXEC regions (v_exp_f32 + fast divide only).
__device__ __forceinline__ float fast_tanh(float x) {
  float e = __expf(2.0f * x);
  return 1.0f - __fdividef(2.0f, e + 1.0f);
}

// A fragment: 16x32 tile. X points at row m=0; row stride ld (halfs); k offset k0.
// Lane m = lane&15, half = lane>>4. halfs[0..15] = K(half*16 .. +15).
// Two contiguous 16B loads.
__device__ __forceinline__ v16h load_a_frag(const _Float16* X, int ld, int k0) {
  int lane = threadIdx.x & 31;
  int m    = lane & 15;
  int h2   = lane >> 4;
  const _Float16* p = X + (size_t)m * ld + k0 + h2 * 16;
  v8h lo = *(const v8h*)p;
  v8h hi = *(const v8h*)(p + 8);
  return __builtin_shufflevector(lo, hi, 0,1,2,3,4,5,6,7,8,9,10,11,12,13,14,15);
}

// B fragment for y = x @ W^T with W row-major (out_features x in_features):
// B[k][n] = W[n][k]. Lane n = n0 + (lane&15), K base = k0 + 16*(lane>>4):
// one contiguous 32B load of W row n.
__device__ __forceinline__ v16h load_b_frag(const _Float16* W, int ld, int n0, int k0) {
  int lane = threadIdx.x & 31;
  int n  = n0 + (lane & 15);
  int kb = k0 + ((lane >> 4) << 4);
  return *(const v16h*)(W + (size_t)n * ld + kb);
}

// ---------------------------------------------------------------------------
// prep kernels
// ---------------------------------------------------------------------------
__global__ void sm_cast_f16(const float* __restrict__ src,
                            _Float16* __restrict__ dst, int n) {
  int stride = gridDim.x * blockDim.x;
  for (int i = blockIdx.x * blockDim.x + threadIdx.x; i < n; i += stride)
    dst[i] = (_Float16)src[i];
}

__global__ void sm_gather_f16(const int* __restrict__ idx,
                              const float* __restrict__ emb,
                              _Float16* __restrict__ dst, int rows) {
  int stride = gridDim.x * blockDim.x;
  int total = rows * EDIM;
  for (int i = blockIdx.x * blockDim.x + threadIdx.x; i < total; i += stride) {
    int r = i >> 9;           // / EDIM
    int e = i & (EDIM - 1);
    dst[i] = (_Float16)emb[(size_t)idx[r] * EDIM + e];
  }
}

// ---------------------------------------------------------------------------
// encoder: 128 sequential tanh-RNN steps, one workgroup (16 waves), h in LDS.
// Each wave owns 4 consecutive N-tiles (64 cols) and reuses one A fragment
// across the 4 B fragments. h_t written (f16) to enc_hid (B,SE,H).
// ---------------------------------------------------------------------------
__global__ __launch_bounds__(RNN_THREADS) void sm_encoder(
    const _Float16* __restrict__ xemb,   // (B,SE,E) f16
    const _Float16* __restrict__ Wih,    // (H,E)  f16
    const _Float16* __restrict__ Whh,    // (H,H)  f16
    const float*    __restrict__ bias,   // (H)
    const float*    __restrict__ state0, // (1,B,H)
    _Float16*       __restrict__ enc_hid)// (B,SE,H) f16
{
  __shared__ _Float16 hsh[BDIM * HDIM];   // 32 KB
  const int tid  = threadIdx.x;
  const int wave = tid >> 5;
  const int lane = tid & 31;
  const int nbase = wave * 64;            // 4 tiles of 16 cols

  for (int i = tid; i < BDIM * HDIM; i += RNN_THREADS)
    hsh[i] = (_Float16)state0[i];
  __syncthreads();

  for (int t = 0; t < SEQE; ++t) {
    v8f acc[4] = {};
    const _Float16* X = xemb + (size_t)t * EDIM;   // row stride SE*E
#pragma unroll 1
    for (int k0 = 0; k0 < EDIM; k0 += 32) {
      v16h a = load_a_frag(X, SEQE * EDIM, k0);
#pragma unroll
      for (int j = 0; j < 4; ++j) {
        v16h b = load_b_frag(Wih, EDIM, nbase + j * 16, k0);
        acc[j] = wmma16(a, b, acc[j]);
      }
    }
#pragma unroll 1
    for (int k0 = 0; k0 < HDIM; k0 += 32) {
      v16h a = load_a_frag(hsh, HDIM, k0);         // ds_load_b128
#pragma unroll
      for (int j = 0; j < 4; ++j) {
        v16h b = load_b_frag(Whh, HDIM, nbase + j * 16, k0);
        acc[j] = wmma16(a, b, acc[j]);
      }
    }
    const int hf = lane >> 4;
#pragma unroll
    for (int j = 0; j < 4; ++j) {
      const int n = nbase + j * 16 + (lane & 15);
      const float bv = bias[n];
#pragma unroll
      for (int r = 0; r < 8; ++r) {
        int m = hf * 8 + r;
        float v = fast_tanh(acc[j][r] + bv);
        enc_hid[((size_t)m * SEQE + t) * HDIM + n] = (_Float16)v;
      }
    }
    __threadfence();
    __syncthreads();
    for (int i = tid; i < BDIM * HDIM; i += RNN_THREADS) {
      int m = i >> 10, n = i & (HDIM - 1);
      hsh[i] = enc_hid[((size_t)m * SEQE + t) * HDIM + n];
    }
    __syncthreads();
  }
}

// ---------------------------------------------------------------------------
// decoder recurrence: 64 sequential steps. Per step:
//   q = h @ mlp_W^T + mlp_b            (WMMA)
//   scores/softmax/ctx                  (VALU, tiny: ~6 MFLOP)
//   h = tanh([e,ctx] @ dec_Wih^T + h @ dec_Whh^T + b)   (WMMA)
// h_t stored f16 into hall (SD,B,H) for the batched output GEMM.
// ---------------------------------------------------------------------------
__global__ __launch_bounds__(RNN_THREADS) void sm_decoder(
    const _Float16* __restrict__ demb,    // (B,SD,E) f16
    const _Float16* __restrict__ enc_hid, // (B,SE,H) f16
    const _Float16* __restrict__ mlpW,    // (H,H)
    const float*    __restrict__ mlpB,
    const _Float16* __restrict__ Wih,     // (H, E+H)
    const _Float16* __restrict__ Whh,     // (H,H)
    const float*    __restrict__ decB,
    _Float16*       __restrict__ qbuf,    // (16,1024) f16 scratch
    _Float16*       __restrict__ inpbuf,  // (16,1536) f16 scratch
    _Float16*       __restrict__ hall,    // (SD,16,1024) f16
    float*          __restrict__ hfin)    // d_out tail: (16,1024) f32
{
  __shared__ _Float16 hsh[BDIM * HDIM];   // 32 KB
  __shared__ float    ssc[BDIM * SEQE];   //  8 KB
  const int tid  = threadIdx.x;
  const int wave = tid >> 5;
  const int lane = tid & 31;
  const int nbase = wave * 64;

  // h0 = last encoder state
  for (int i = tid; i < BDIM * HDIM; i += RNN_THREADS) {
    int m = i >> 10, n = i & (HDIM - 1);
    hsh[i] = enc_hid[((size_t)m * SEQE + (SEQE - 1)) * HDIM + n];
  }
  __syncthreads();

  for (int t = 0; t < SEQD; ++t) {
    // ---- phase 1: q = h @ mlp_W^T + mlp_b ----
    {
      v8f acc[4] = {};
#pragma unroll 1
      for (int k0 = 0; k0 < HDIM; k0 += 32) {
        v16h a = load_a_frag(hsh, HDIM, k0);
#pragma unroll
        for (int j = 0; j < 4; ++j) {
          v16h b = load_b_frag(mlpW, HDIM, nbase + j * 16, k0);
          acc[j] = wmma16(a, b, acc[j]);
        }
      }
      const int hf = lane >> 4;
#pragma unroll
      for (int j = 0; j < 4; ++j) {
        const int n = nbase + j * 16 + (lane & 15);
        const float bv = mlpB[n];
#pragma unroll
        for (int r = 0; r < 8; ++r)
          qbuf[(hf * 8 + r) * HDIM + n] = (_Float16)(acc[j][r] + bv);
      }
    }
    __threadfence();
    __syncthreads();

    // ---- phase 2: scores[b][s] = enc_hid[b][s] . q[b]  (16B vector loads) ----
    for (int p = tid; p < BDIM * SEQE; p += RNN_THREADS) {
      int b = p >> 7, s = p & (SEQE - 1);
      const _Float16* eh = enc_hid + ((size_t)b * SEQE + s) * HDIM;
      const _Float16* qq = qbuf + b * HDIM;
      float a = 0.f;
#pragma unroll 1
      for (int k = 0; k < HDIM; k += 8) {
        v8h ev = *(const v8h*)(eh + k);
        v8h qv = *(const v8h*)(qq + k);
#pragma unroll
        for (int u = 0; u < 8; ++u) a += (float)ev[u] * (float)qv[u];
      }
      ssc[b * SEQE + s] = a;
    }
    __syncthreads();

    // ---- phase 3: softmax over s (16 rows of 128) ----
    if (tid < BDIM) {
      float mx = -1e30f;
      for (int s = 0; s < SEQE; ++s) mx = fmaxf(mx, ssc[tid * SEQE + s]);
      float sum = 0.f;
      for (int s = 0; s < SEQE; ++s) {
        float e = __expf(ssc[tid * SEQE + s] - mx);
        ssc[tid * SEQE + s] = e;
        sum += e;
      }
      float inv = 1.0f / sum;
      for (int s = 0; s < SEQE; ++s) ssc[tid * SEQE + s] *= inv;
    }
    __syncthreads();

    // ---- phase 4: ctx + embedding -> inp = [e, ctx] (16 x 1536) ----
    for (int hh = tid; hh < HDIM; hh += RNN_THREADS) {
      for (int b = 0; b < BDIM; ++b) {
        const _Float16* eh = enc_hid + (size_t)b * SEQE * HDIM + hh;
        float a = 0.f;
        for (int s = 0; s < SEQE; ++s) a += ssc[b * SEQE + s] * (float)eh[(size_t)s * HDIM];
        inpbuf[b * IDIM + EDIM + hh] = (_Float16)a;
      }
    }
    for (int i = tid; i < BDIM * EDIM; i += RNN_THREADS) {
      int b = i >> 9, e = i & (EDIM - 1);
      inpbuf[b * IDIM + e] = demb[((size_t)b * SEQD + t) * EDIM + e];
    }
    __threadfence();
    __syncthreads();

    // ---- phase 5: h = tanh(inp @ dec_Wih^T + h @ dec_Whh^T + b) ----
    {
      v8f acc[4] = {};
#pragma unroll 1
      for (int k0 = 0; k0 < IDIM; k0 += 32) {
        v16h a = load_a_frag(inpbuf, IDIM, k0);
#pragma unroll
        for (int j = 0; j < 4; ++j) {
          v16h b = load_b_frag(Wih, IDIM, nbase + j * 16, k0);
          acc[j] = wmma16(a, b, acc[j]);
        }
      }
#pragma unroll 1
      for (int k0 = 0; k0 < HDIM; k0 += 32) {
        v16h a = load_a_frag(hsh, HDIM, k0);
#pragma unroll
        for (int j = 0; j < 4; ++j) {
          v16h b = load_b_frag(Whh, HDIM, nbase + j * 16, k0);
          acc[j] = wmma16(a, b, acc[j]);
        }
      }
      const int hf = lane >> 4;
#pragma unroll
      for (int j = 0; j < 4; ++j) {
        const int n = nbase + j * 16 + (lane & 15);
        const float bv = decB[n];
#pragma unroll
        for (int r = 0; r < 8; ++r) {
          int m = hf * 8 + r;
          float v = fast_tanh(acc[j][r] + bv);
          hall[((size_t)t * BDIM + m) * HDIM + n] = (_Float16)v;
          if (t == SEQD - 1) hfin[m * HDIM + n] = v;
        }
      }
    }
    __threadfence();
    __syncthreads();
    for (int i = tid; i < BDIM * HDIM; i += RNN_THREADS) {
      int m = i >> 10, n = i & (HDIM - 1);
      hsh[i] = hall[((size_t)t * BDIM + m) * HDIM + n];
    }
    __syncthreads();
  }
}

// ---------------------------------------------------------------------------
// batched output projection: (B*SD=1024, O=32000, K=1024) WMMA GEMM.
// Wave register-blocks 4 M-tiles (t) x 2 N-tiles; out_W f16 stays L2-resident.
// 256 threads = 8 waves -> 2 waves/SIMD, ~140 live VGPRs fits without spill.
// ---------------------------------------------------------------------------
__global__ __launch_bounds__(256) void sm_outproj(
    const _Float16* __restrict__ hall,   // (SD,16,1024) f16
    const _Float16* __restrict__ outW,   // (O,H) f16
    const float*    __restrict__ outB,   // (O)
    float*          __restrict__ out)    // (B,SD,O) f32
{
  const int wave = threadIdx.x >> 5;
  const int lane = threadIdx.x & 31;
  const int n0 = (blockIdx.x * 8 + wave) * 32;   // 2 N-tiles
  const int t0 = blockIdx.y * 4;                 // 4 M-tiles

  v8f acc[4][2] = {};
#pragma unroll 1
  for (int k0 = 0; k0 < HDIM; k0 += 32) {
    // hint the next out_W chunk toward the caches
    __builtin_prefetch(outW + (size_t)(n0 + (lane & 15)) * HDIM + k0 + 64, 0, 1);
    v16h a0 = load_a_frag(hall + (size_t)(t0 + 0) * BDIM * HDIM, HDIM, k0);
    v16h a1 = load_a_frag(hall + (size_t)(t0 + 1) * BDIM * HDIM, HDIM, k0);
    v16h a2 = load_a_frag(hall + (size_t)(t0 + 2) * BDIM * HDIM, HDIM, k0);
    v16h a3 = load_a_frag(hall + (size_t)(t0 + 3) * BDIM * HDIM, HDIM, k0);
    v16h b0 = load_b_frag(outW, HDIM, n0,      k0);
    v16h b1 = load_b_frag(outW, HDIM, n0 + 16, k0);
    acc[0][0] = wmma16(a0, b0, acc[0][0]);
    acc[0][1] = wmma16(a0, b1, acc[0][1]);
    acc[1][0] = wmma16(a1, b0, acc[1][0]);
    acc[1][1] = wmma16(a1, b1, acc[1][1]);
    acc[2][0] = wmma16(a2, b0, acc[2][0]);
    acc[2][1] = wmma16(a2, b1, acc[2][1]);
    acc[3][0] = wmma16(a3, b0, acc[3][0]);
    acc[3][1] = wmma16(a3, b1, acc[3][1]);
  }

  const int nl = lane & 15;
  const int hf = lane >> 4;
#pragma unroll
  for (int i = 0; i < 4; ++i) {
#pragma unroll
    for (int j = 0; j < 2; ++j) {
      const int n = n0 + j * 16 + nl;
      const float bv = outB[n];
      const int t = t0 + i;
#pragma unroll
      for (int r = 0; r < 8; ++r) {
        int m = hf * 8 + r;                                  // = batch b
        out[((size_t)m * SEQD + t) * ODIM + n] = acc[i][j][r] + bv;
      }
    }
  }
}

// ---------------------------------------------------------------------------
// launch
// ---------------------------------------------------------------------------
extern "C" void kernel_launch(void* const* d_in, const int* in_sizes, int n_in,
                              void* d_out, int out_size, void* d_ws, size_t ws_size,
                              hipStream_t stream) {
  (void)in_sizes; (void)n_in; (void)out_size; (void)ws_size;
  const int*   enc_x  = (const int*)  d_in[0];
  const int*   dec_y  = (const int*)  d_in[1];
  const float* state0 = (const float*)d_in[2];
  const float* emb    = (const float*)d_in[3];
  const float* encWih = (const float*)d_in[4];
  const float* encWhh = (const float*)d_in[5];
  const float* enc_b  = (const float*)d_in[6];
  const float* mlp_W  = (const float*)d_in[7];
  const float* mlp_b  = (const float*)d_in[8];
  const float* decWih = (const float*)d_in[9];
  const float* decWhh = (const float*)d_in[10];
  const float* dec_b  = (const float*)d_in[11];
  const float* out_W  = (const float*)d_in[12];
  const float* out_b  = (const float*)d_in[13];
  float* out = (float*)d_out;
  _Float16* ws = (_Float16*)d_ws;

  // workspace offsets (in halfs)
  constexpr size_t OFF_EWIH = 0;                                 // 512*1024
  constexpr size_t OFF_EWHH = OFF_EWIH + (size_t)HDIM * EDIM;    // 1024*1024
  constexpr size_t OFF_MLPW = OFF_EWHH + (size_t)HDIM * HDIM;
  constexpr size_t OFF_DWIH = OFF_MLPW + (size_t)HDIM * HDIM;    // 1024*1536
  constexpr size_t OFF_DWHH = OFF_DWIH + (size_t)HDIM * IDIM;
  constexpr size_t OFF_OUTW = OFF_DWHH + (size_t)HDIM * HDIM;    // 32000*1024
  constexpr size_t OFF_EEMB = OFF_OUTW + (size_t)ODIM * HDIM;    // 16*128*512
  constexpr size_t OFF_DEMB = OFF_EEMB + (size_t)BDIM * SEQE * EDIM;
  constexpr size_t OFF_EHID = OFF_DEMB + (size_t)BDIM * SEQD * EDIM;
  constexpr size_t OFF_Q    = OFF_EHID + (size_t)BDIM * SEQE * HDIM;
  constexpr size_t OFF_INP  = OFF_Q    + (size_t)BDIM * HDIM;
  constexpr size_t OFF_HALL = OFF_INP  + (size_t)BDIM * IDIM;

  auto cast = [&](const float* s, size_t off, int n) {
    int blocks = (n + 255) / 256; if (blocks > 8192) blocks = 8192;
    sm_cast_f16<<<blocks, 256, 0, stream>>>(s, ws + off, n);
  };
  cast(encWih, OFF_EWIH, HDIM * EDIM);
  cast(encWhh, OFF_EWHH, HDIM * HDIM);
  cast(mlp_W,  OFF_MLPW, HDIM * HDIM);
  cast(decWih, OFF_DWIH, HDIM * IDIM);
  cast(decWhh, OFF_DWHH, HDIM * HDIM);
  cast(out_W,  OFF_OUTW, ODIM * HDIM);

  sm_gather_f16<<<2048, 256, 0, stream>>>(enc_x, emb, ws + OFF_EEMB, BDIM * SEQE);
  sm_gather_f16<<<1024, 256, 0, stream>>>(dec_y, emb, ws + OFF_DEMB, BDIM * SEQD);

  sm_encoder<<<1, RNN_THREADS, 0, stream>>>(ws + OFF_EEMB, ws + OFF_EWIH,
                                            ws + OFF_EWHH, enc_b, state0,
                                            ws + OFF_EHID);

  // h_fin goes straight after the (B,SD,O) logits block in d_out
  float* hfin = out + (size_t)BDIM * SEQD * ODIM;
  sm_decoder<<<1, RNN_THREADS, 0, stream>>>(ws + OFF_DEMB, ws + OFF_EHID,
                                            ws + OFF_MLPW, mlp_b,
                                            ws + OFF_DWIH, ws + OFF_DWHH, dec_b,
                                            ws + OFF_Q, ws + OFF_INP,
                                            ws + OFF_HALL, hfin);

  sm_outproj<<<dim3(125, 16), 256, 0, stream>>>(ws + OFF_HALL, ws + OFF_OUTW,
                                                out_b, out);
}
